// OptimizedSimEncoderLayer_37383395344636
// MI455X (gfx1250) — matrix operand order repaired and checked by reading, not compiled, not run on hardware
//
#include <hip/hip_runtime.h>
#include <hip/hip_bf16.h>
#include <math.h>

// ---------------------------------------------------------------------------
// MI455X (gfx1250) implementation of the cached ViT encoder layer.
//
// Only the class-token (n==0) row feeds the output, so Q projection,
// attention and the tail are pruned 197x. Bulk compute = K/V projections +
// diff-MLP as bf16 WMMA GEMMs (v_wmma_f32_16x16x32_bf16, wave32), with tiles
// staged via GLOBAL_LOAD_ASYNC_TO_LDS_B128 (ASYNCcnt) when the toolchain
// exposes the builtin. ~35 GFLOP vs ~200MB traffic -> memory-bound at
// 23.3 TB/s; WMMA + async staging keeps the pipeline bandwidth-limited.
// ---------------------------------------------------------------------------

#define Bsz   16
#define Nseq  197
#define DIM   768
#define Hh    12
#define HD    64
#define SCALE 0.125f          // HD^-0.5
#define Trows 12608           // 4*B*N
#define CACHE_ROWS 3152       // B*N
#define INNER 64
#define FF    3072
#define FB    64              // 4*B

typedef __attribute__((ext_vector_type(16))) __bf16 v16bf;
typedef __attribute__((ext_vector_type(8)))  float  v8f;

// Async global->LDS staging (gfx1250). Guarded so the TU compiles on
// toolchains without the builtin; histogram shows which path lowered.
#if defined(__HIP_DEVICE_COMPILE__) && defined(__has_builtin)
# if __has_builtin(__builtin_amdgcn_global_load_async_to_lds_b128) && \
     __has_builtin(__builtin_amdgcn_s_wait_asynccnt)
#  define USE_ASYNC_LDS 1
# endif
#endif
#ifndef USE_ASYNC_LDS
# define USE_ASYNC_LDS 0
#endif

#if USE_ASYNC_LDS
typedef int v4i_async __attribute__((vector_size(16)));  // clang prints param as int __vector(4)
typedef __attribute__((address_space(1))) v4i_async* gas_v4i;
typedef __attribute__((address_space(3))) v4i_async* las_v4i;
#endif

__device__ __forceinline__ void stage16B(const unsigned short* g, unsigned short* l) {
#if USE_ASYNC_LDS
  __builtin_amdgcn_global_load_async_to_lds_b128(
      (gas_v4i)(g), (las_v4i)(l), 0, 0);
#else
  *(uint4*)l = *(const uint4*)g;
#endif
}
__device__ __forceinline__ void stage_fence() {
#if USE_ASYNC_LDS
  __builtin_amdgcn_s_wait_asynccnt(0);
#endif
}

__device__ __forceinline__ unsigned short f32_to_bf16(float f) {
  unsigned int u = __float_as_uint(f);
  u += 0x7FFFu + ((u >> 16) & 1u);   // round-to-nearest-even
  return (unsigned short)(u >> 16);
}
__device__ __forceinline__ float bf16_to_f32(unsigned short h) {
  return __uint_as_float(((unsigned int)h) << 16);
}

// ------------------------------- LN1 + cast --------------------------------
__global__ __launch_bounds__(256)
void ln1_cast_kernel(const float* __restrict__ hs, const float* __restrict__ g,
                     const float* __restrict__ b, unsigned short* __restrict__ xout) {
  __shared__ float s1[256], s2[256];
  const int row = blockIdx.x, tid = threadIdx.x;
  const float* r = hs + (size_t)row * DIM;
  float v0 = r[tid], v1 = r[tid + 256], v2 = r[tid + 512];
  s1[tid] = v0 + v1 + v2;
  s2[tid] = v0 * v0 + v1 * v1 + v2 * v2;
  __syncthreads();
  for (int off = 128; off > 0; off >>= 1) {
    if (tid < off) { s1[tid] += s1[tid + off]; s2[tid] += s2[tid + off]; }
    __syncthreads();
  }
  const float mean = s1[0] * (1.0f / DIM);
  const float var  = s2[0] * (1.0f / DIM) - mean * mean;
  const float rstd = rsqrtf(var + 1e-5f);
  unsigned short* o = xout + (size_t)row * DIM;
  #pragma unroll
  for (int i = 0; i < 3; i++) {
    int c = tid + i * 256;
    float x = (r[c] - mean) * rstd * g[c] + b[c];
    o[c] = f32_to_bf16(x);
  }
}

// -------------------- casts / transposed weight packing --------------------
__global__ void cast_bf16_kernel(const float* __restrict__ in,
                                 unsigned short* __restrict__ out, size_t n) {
  for (size_t i = blockIdx.x * 256ull + threadIdx.x; i < n; i += (size_t)gridDim.x * 256ull)
    out[i] = f32_to_bf16(in[i]);
}
// wkvT[c][k] = c<768 ? wk[k][c] : wv[k][c-768]      (N-major: K contiguous)
__global__ void pack_wkvT_kernel(const float* __restrict__ wk, const float* __restrict__ wv,
                                 unsigned short* __restrict__ out) {
  for (size_t i = blockIdx.x * 256ull + threadIdx.x; i < (size_t)1536 * DIM;
       i += (size_t)gridDim.x * 256ull) {
    size_t c = i / DIM, k = i % DIM;
    float v = (c < DIM) ? wk[k * DIM + c] : wv[k * DIM + (c - DIM)];
    out[i] = f32_to_bf16(v);
  }
}
// w1T[c][k] = rest_w1[k][c]
__global__ void pack_w1T_kernel(const float* __restrict__ w1, unsigned short* __restrict__ out) {
  for (size_t i = blockIdx.x * 256ull + threadIdx.x; i < (size_t)INNER * DIM;
       i += (size_t)gridDim.x * 256ull) {
    size_t c = i / DIM, k = i % DIM;
    out[i] = f32_to_bf16(w1[k * INNER + c]);
  }
}
// w2kvT[c][j] = rest_w2[j][1536+c]   (only the kd/vd half is needed densely)
__global__ void pack_w2kvT_kernel(const float* __restrict__ w2, unsigned short* __restrict__ out) {
  for (size_t i = blockIdx.x * 256ull + threadIdx.x; i < (size_t)1536 * INNER;
       i += (size_t)gridDim.x * 256ull) {
    size_t c = i / INNER, j = i % INNER;
    out[i] = f32_to_bf16(w2[j * FF + 1536 + c]);
  }
}
__global__ void pack_biaskv_kernel(const float* __restrict__ bk, const float* __restrict__ bv,
                                   float* __restrict__ out) {
  int i = blockIdx.x * 256 + threadIdx.x;
  if (i < 1536) out[i] = (i < DIM) ? bk[i] : bv[i - DIM];
}

// ------------------------- bf16 WMMA tiled GEMM ----------------------------
// C[M,Nn] = A[M,K](bf16, row-major) * BT[Nn,K](bf16, N-major) (+bias)
// EPI: 0=none 1=tanh-GELU. 8 waves/block, 16x16x32 bf16 WMMA fragments.
// Both A and B tiles are K-contiguous -> pure 16B async LDS copies.
template<int BM, int BN, int WM, int WN, int EPI, int OUTBF16>
__global__ __launch_bounds__(256)
void gemm_bf16_wmma(const unsigned short* __restrict__ A,
                    const unsigned short* __restrict__ BT,
                    const float* __restrict__ bias,
                    void* __restrict__ C, int Mrows, int K, int Nn) {
  constexpr int LDT = 40;                 // ushort LDS stride (80B: 16B-aligned, bank-skewed)
  __shared__ unsigned short Alds[BM * LDT];
  __shared__ unsigned short Blds[BN * LDT];   // Blds[n][k]
  const int tid  = threadIdx.x;
  const int lane = tid & 31;
  const int wave = tid >> 5;
  constexpr int WAVES_M = BM / WM;
  constexpr int WAVES_N = BN / WN;
  static_assert(WAVES_M * WAVES_N == 8, "8 waves per block");
  constexpr int FM = WM / 16, FN = WN / 16;
  const int wm = (wave % WAVES_M) * WM;
  const int wn = (wave / WAVES_M) * WN;
  const int m0 = blockIdx.x * BM;
  const int n0 = blockIdx.y * BN;
  const int lrow = lane & 15;
  const int akb  = (lane >> 4) << 3;      // A frag K base: 0 or 8
  const int bkb  = (lane >> 4) << 3;      // B frag K base: 0 or 8

  v8f acc[FM][FN] = {};

  for (int k0 = 0; k0 < K; k0 += 32) {
    // ---- stage A tile: BM x 32 bf16, 16B chunks, async -> LDS
    for (int c = tid; c < BM * 4; c += 256) {
      int row = c >> 2, ko = (c & 3) << 3;
      int gr = m0 + row;
      if (gr < Mrows)
        stage16B(A + (size_t)gr * K + k0 + ko, &Alds[row * LDT + ko]);
      else
        *(uint4*)(&Alds[row * LDT + ko]) = make_uint4(0u, 0u, 0u, 0u);
    }
    // ---- stage B tile: BN x 32 bf16 (pre-transposed in global), async -> LDS
    for (int c = tid; c < BN * 4; c += 256) {
      int n = c >> 2, ko = (c & 3) << 3;
      stage16B(BT + (size_t)(n0 + n) * K + k0 + ko, &Blds[n * LDT + ko]);
    }
    stage_fence();
    __syncthreads();

    // prefetch next K tile of A (global_prefetch_b8 on gfx1250)
    if (k0 + 32 < K && tid < BM * 4) {
      int row = tid >> 2, ko = (tid & 3) << 3;
      int gr = m0 + row;
      if (gr < Mrows) __builtin_prefetch(A + (size_t)gr * K + k0 + 32 + ko, 0, 1);
    }

    // ---- build fragments per ISA VGPR layout
    v16bf afr[FM], bfr[FN];
    #pragma unroll
    for (int i = 0; i < FM; i++) {
      // A 16x32: lanes 0-15 K {0..7,16..23}; lanes 16-31 K {8..15,24..31}
      const unsigned short* p = &Alds[(wm + i * 16 + lrow) * LDT + akb];
      uint4* pv = (uint4*)&afr[i];
      pv[0] = *(const uint4*)p;
      pv[1] = *(const uint4*)(p + 16);
    }
    #pragma unroll
    for (int i = 0; i < FN; i++) {
      // B 32x16: lanes 0-15 K {0..7,16..23}; lanes 16-31 K {8..15,24..31}
      const unsigned short* p = &Blds[(wn + i * 16 + lrow) * LDT + bkb];
      uint4* pv = (uint4*)&bfr[i];
      pv[0] = *(const uint4*)p;
      pv[1] = *(const uint4*)(p + 16);
    }
    #pragma unroll
    for (int i = 0; i < FM; i++)
      #pragma unroll
      for (int j = 0; j < FN; j++)
        acc[i][j] = __builtin_amdgcn_wmma_f32_16x16x32_bf16(
            false, afr[i], false, bfr[j], (short)0, acc[i][j], false, false);
    __syncthreads();
  }

  // ---- epilogue: C layout lane=n+16*(m/8), vgpr=m%8
  const int mb = (lane >> 4) << 3;
  #pragma unroll
  for (int i = 0; i < FM; i++)
    #pragma unroll
    for (int j = 0; j < FN; j++) {
      const int col = n0 + wn + j * 16 + lrow;
      const float bvv = bias ? bias[col] : 0.0f;
      #pragma unroll
      for (int e = 0; e < 8; e++) {
        const int grow = m0 + wm + i * 16 + mb + e;
        if (grow < Mrows) {
          float x = acc[i][j][e] + bvv;
          if (EPI == 1) {  // tanh-approx GELU (jax.nn.gelu default)
            float t = tanhf(0.7978845608f * (x + 0.044715f * x * x * x));
            x = 0.5f * x * (1.0f + t);
          }
          if (OUTBF16)
            ((unsigned short*)C)[(size_t)grow * Nn + col] = f32_to_bf16(x);
          else
            ((float*)C)[(size_t)grow * Nn + col] = x;
        }
      }
    }
}

// -------------------- CLS h/q assembly (64 rows only) ----------------------
__global__ __launch_bounds__(256)
void cls_kernel(const float* __restrict__ hs, const float* __restrict__ hqkv,
                const unsigned short* __restrict__ x_bf,
                const float* __restrict__ wq, const float* __restrict__ bq,
                const unsigned short* __restrict__ g1_bf,
                const float* __restrict__ rest_w2,
                const int* __restrict__ gidx, const unsigned char* __restrict__ reuse,
                float* __restrict__ h_cls, float* __restrict__ q_cls) {
  const int fb = blockIdx.x, tid = threadIdx.x;
  __shared__ float xr[DIM];
  __shared__ float g1r[INNER];
  __shared__ int   sidx;
  __shared__ int   smk;
  if (tid == 0) { sidx = gidx[fb * Nseq]; smk = reuse[fb * Nseq]; }
  __syncthreads();
  const int idx = sidx;
  const bool m = smk != 0;
  const bool fromCache = idx < CACHE_ROWS;
  if (!fromCache) {
    const int r = idx - CACHE_ROWS;
    for (int c = tid; c < DIM; c += 256) xr[c] = bf16_to_f32(x_bf[(size_t)r * DIM + c]);
  }
  if (tid < INNER) g1r[tid] = bf16_to_f32(g1_bf[(size_t)(fb * Nseq) * INNER + tid]);
  __syncthreads();
  for (int c = tid; c < DIM; c += 256) {
    float hv = fromCache ? hqkv[(size_t)idx * DIM + c]                      // h_cache slice 0
                         : hs[(size_t)(idx - CACHE_ROWS) * DIM + c];
    float qv;
    if (fromCache) {
      qv = hqkv[(size_t)(CACHE_ROWS + idx) * DIM + c];                      // q_cache slice 1
    } else {
      float a = bq[c];
      for (int k = 0; k < DIM; k++) a += xr[k] * wq[(size_t)k * DIM + c];
      qv = a;
    }
    if (m) {
      float hd = 0.f, qd = 0.f;
      for (int j = 0; j < INNER; j++) {
        const float gj = g1r[j];
        hd += gj * rest_w2[(size_t)j * FF + c];
        qd += gj * rest_w2[(size_t)j * FF + DIM + c];
      }
      hv += hd; qv += qd;
    }
    h_cls[fb * DIM + c] = hv;
    q_cls[fb * DIM + c] = qv * SCALE;
  }
}

// -------------- attention: 1 CLS query x 197 keys per (fb,h) --------------
__global__ __launch_bounds__(256)
void attn_kernel(const float* __restrict__ q_cls, const int* __restrict__ gidx,
                 const unsigned char* __restrict__ reuse,
                 const float* __restrict__ hqkv, const float* __restrict__ kv_new,
                 const float* __restrict__ kvdiff, float* __restrict__ attn_out) {
  const int h = blockIdx.x, fb = blockIdx.y, tid = threadIdx.x;
  __shared__ float qv[HD];
  __shared__ float sc[256];
  __shared__ int   sidx[208];
  __shared__ unsigned char smk[208];
  __shared__ float red[256];
  __shared__ float bc[2];
  if (tid < HD) qv[tid] = q_cls[fb * DIM + h * HD + tid];
  if (tid < Nseq) { sidx[tid] = gidx[fb * Nseq + tid]; smk[tid] = reuse[fb * Nseq + tid]; }
  __syncthreads();
  float s = -3.0e38f;
  if (tid < Nseq) {
    const int idx = sidx[tid];
    const float* kb = (idx < CACHE_ROWS)
        ? hqkv + ((size_t)2 * CACHE_ROWS + idx) * DIM + h * HD                // k_cache slice 2
        : kv_new + (size_t)(idx - CACHE_ROWS) * 1536 + h * HD;
    const float* kd = kvdiff + (size_t)(fb * Nseq + tid) * 1536 + h * HD;
    const bool m = smk[tid] != 0;
    float a = 0.f;
    for (int d = 0; d < HD; d++) {
      float kk = kb[d];
      if (m) kk += kd[d];
      a += qv[d] * kk;
    }
    s = a;
  }
  red[tid] = s; __syncthreads();
  for (int off = 128; off > 0; off >>= 1) {
    if (tid < off) red[tid] = fmaxf(red[tid], red[tid + off]);
    __syncthreads();
  }
  if (tid == 0) bc[0] = red[0];
  __syncthreads();
  const float p = (tid < Nseq) ? __expf(s - bc[0]) : 0.0f;
  sc[tid] = p; red[tid] = p; __syncthreads();
  for (int off = 128; off > 0; off >>= 1) {
    if (tid < off) red[tid] += red[tid + off];
    __syncthreads();
  }
  if (tid == 0) bc[1] = red[0];
  __syncthreads();
  if (tid < HD) {
    const int d = tid;
    float a = 0.f;
    for (int n = 0; n < Nseq; n++) {
      const int idx = sidx[n];
      float vv = (idx < CACHE_ROWS)
          ? hqkv[((size_t)3 * CACHE_ROWS + idx) * DIM + h * HD + d]           // v_cache slice 3
          : kv_new[(size_t)(idx - CACHE_ROWS) * 1536 + 768 + h * HD + d];
      if (smk[n]) vv += kvdiff[(size_t)(fb * Nseq + n) * 1536 + 768 + h * HD + d];
      a += sc[n] * vv;
    }
    attn_out[fb * DIM + h * HD + d] = a / bc[1];
  }
}

// ----------------- tail: wo + LN2 + quickGELU MLP (64 rows) ----------------
__global__ __launch_bounds__(256)
void tail_kernel(const float* __restrict__ h_cls, const float* __restrict__ attn_out,
                 const float* __restrict__ wo, const float* __restrict__ bo,
                 const float* __restrict__ g2, const float* __restrict__ b2,
                 const float* __restrict__ fc1w, const float* __restrict__ fc1b,
                 const float* __restrict__ fc2w, const float* __restrict__ fc2b,
                 float* __restrict__ out) {
  const int fb = blockIdx.x, tid = threadIdx.x;
  __shared__ float ain[DIM];
  __shared__ float resid[DIM];
  __shared__ float vln[DIM];
  __shared__ float t1[FF];
  __shared__ float red1[256], red2[256];
  for (int c = tid; c < DIM; c += 256) ain[c] = attn_out[fb * DIM + c];
  __syncthreads();
  for (int c = tid; c < DIM; c += 256) {
    float a = bo[c];
    for (int k = 0; k < DIM; k++) a += ain[k] * wo[(size_t)k * DIM + c];
    resid[c] = h_cls[fb * DIM + c] + a;
  }
  __syncthreads();
  { // LN2
    float p1 = 0.f, p2 = 0.f;
    for (int c = tid; c < DIM; c += 256) { float v = resid[c]; p1 += v; p2 += v * v; }
    red1[tid] = p1; red2[tid] = p2; __syncthreads();
    for (int off = 128; off > 0; off >>= 1) {
      if (tid < off) { red1[tid] += red1[tid + off]; red2[tid] += red2[tid + off]; }
      __syncthreads();
    }
    const float mean = red1[0] * (1.0f / DIM);
    const float var  = red2[0] * (1.0f / DIM) - mean * mean;
    const float rstd = rsqrtf(var + 1e-5f);
    for (int c = tid; c < DIM; c += 256)
      vln[c] = (resid[c] - mean) * rstd * g2[c] + b2[c];
  }
  __syncthreads();
  for (int cc = tid; cc < FF; cc += 256) {
    float a = fc1b[cc];
    for (int k = 0; k < DIM; k++) a += vln[k] * fc1w[(size_t)k * FF + cc];
    t1[cc] = a * (1.0f / (1.0f + __expf(-1.702f * a)));   // quick-GELU
  }
  __syncthreads();
  for (int c = tid; c < DIM; c += 256) {
    float a = fc2b[c];
    for (int k = 0; k < FF; k++) a += t1[k] * fc2w[(size_t)k * DIM + c];
    out[fb * DIM + c] = resid[c] + a;
  }
}

// ---------------------------------------------------------------------------
extern "C" void kernel_launch(void* const* d_in, const int* in_sizes, int n_in,
                              void* d_out, int out_size, void* d_ws, size_t ws_size,
                              hipStream_t stream) {
  const float* hs     = (const float*)d_in[0];
  const float* dpp    = (const float*)d_in[1];
  const float* hqkv   = (const float*)d_in[2];
  const float* ln1_g  = (const float*)d_in[3];
  const float* ln1_b  = (const float*)d_in[4];
  const float* wq     = (const float*)d_in[5];
  const float* bq     = (const float*)d_in[6];
  const float* wk     = (const float*)d_in[7];
  const float* bk     = (const float*)d_in[8];
  const float* wv     = (const float*)d_in[9];
  const float* bv     = (const float*)d_in[10];
  const float* rw1    = (const float*)d_in[11];
  const float* rw2    = (const float*)d_in[12];
  const float* wo     = (const float*)d_in[13];
  const float* bo     = (const float*)d_in[14];
  const float* ln2_g  = (const float*)d_in[15];
  const float* ln2_b  = (const float*)d_in[16];
  const float* fc1w   = (const float*)d_in[17];
  const float* fc1b   = (const float*)d_in[18];
  const float* fc2w   = (const float*)d_in[19];
  const float* fc2b   = (const float*)d_in[20];
  const int*   gidx   = (const int*)d_in[21];
  const unsigned char* reuse = (const unsigned char*)d_in[22];  // jnp bool_ = 1 byte

  char* ws = (char*)d_ws;
  size_t off = 0;
  auto alloc = [&](size_t bytes) -> char* {
    char* p = ws + off;
    off = (off + bytes + 255) & ~(size_t)255;
    return p;
  };
  unsigned short* x_bf    = (unsigned short*)alloc((size_t)Trows * DIM * 2);     // LN1(hs) bf16
  unsigned short* dpp_bf  = (unsigned short*)alloc((size_t)Trows * DIM * 2);
  unsigned short* wkvT_bf = (unsigned short*)alloc((size_t)1536 * DIM * 2);      // [wk|wv]^T
  unsigned short* w1T_bf  = (unsigned short*)alloc((size_t)INNER * DIM * 2);     // rw1^T
  unsigned short* w2kvT_bf= (unsigned short*)alloc((size_t)1536 * INNER * 2);    // rw2[:,1536:]^T
  float*          biaskv  = (float*)alloc(1536 * 4);
  float*          kv_new  = (float*)alloc((size_t)Trows * 1536 * 4);             // [k_new|v_new]
  unsigned short* g1_bf   = (unsigned short*)alloc((size_t)Trows * INNER * 2);   // gelu(dpp@rw1)
  float*          kvdiff  = (float*)alloc((size_t)Trows * 1536 * 4);             // [kd|vd]
  float*          h_cls   = (float*)alloc((size_t)FB * DIM * 4);
  float*          q_cls   = (float*)alloc((size_t)FB * DIM * 4);
  float*          att_o   = (float*)alloc((size_t)FB * DIM * 4);

  // 1) prep / casts (weights packed N-major so GEMM tiles are K-contiguous)
  ln1_cast_kernel<<<Trows, 256, 0, stream>>>(hs, ln1_g, ln1_b, x_bf);
  cast_bf16_kernel<<<2048, 256, 0, stream>>>(dpp, dpp_bf, (size_t)Trows * DIM);
  pack_wkvT_kernel<<<1024, 256, 0, stream>>>(wk, wv, wkvT_bf);
  pack_w1T_kernel<<<192, 256, 0, stream>>>(rw1, w1T_bf);
  pack_w2kvT_kernel<<<384, 256, 0, stream>>>(rw2, w2kvT_bf);
  pack_biaskv_kernel<<<6, 256, 0, stream>>>(bk, bv, biaskv);

  // 2) kv_new = LN1(hs) @ [wk|wv] + [bk|bv]      (12608 x 768 x 1536, WMMA)
  {
    dim3 grid((Trows + 127) / 128, 1536 / 128);
    gemm_bf16_wmma<128, 128, 32, 64, 0, 0><<<grid, 256, 0, stream>>>(
        x_bf, wkvT_bf, biaskv, kv_new, Trows, DIM, 1536);
  }
  // 3) g1 = tanh-GELU(dpp @ rest_w1)             (12608 x 768 x 64, WMMA, bf16 out)
  {
    dim3 grid((Trows + 127) / 128, 1);
    gemm_bf16_wmma<128, 64, 32, 32, 1, 1><<<grid, 256, 0, stream>>>(
        dpp_bf, w1T_bf, nullptr, g1_bf, Trows, DIM, INNER);
  }
  // 4) kvdiff = g1 @ rest_w2[:,1536:3072]        (12608 x 64 x 1536, WMMA)
  {
    dim3 grid((Trows + 127) / 128, 1536 / 128);
    gemm_bf16_wmma<128, 128, 32, 64, 0, 0><<<grid, 256, 0, stream>>>(
        g1_bf, w2kvT_bf, nullptr, kvdiff, Trows, INNER, 1536);
  }
  // 5) CLS h/q rows (64), on-demand q projection + hd/qd diff
  cls_kernel<<<FB, 256, 0, stream>>>(hs, hqkv, x_bf, wq, bq, g1_bf, rw2,
                                     gidx, reuse, h_cls, q_cls);
  // 6) attention: CLS query only, gather + diff applied on the fly
  {
    dim3 grid(Hh, FB);
    attn_kernel<<<grid, 256, 0, stream>>>(q_cls, gidx, reuse, hqkv, kv_new,
                                          kvdiff, att_o);
  }
  // 7) wo + LN2 + quickGELU MLP + residual -> d_out (4,B,1,768)
  tail_kernel<<<FB, 256, 0, stream>>>(h_cls, att_o, wo, bo, ln2_g, ln2_b,
                                      fc1w, fc1b, fc2w, fc2b, (float*)d_out);
  (void)in_sizes; (void)n_in; (void)out_size; (void)ws_size;
}
